// DBOT_Sinkhorn_Loss_86311662780618
// MI455X (gfx1250) — matrix operand order repaired and checked by reading, not compiled, not run on hardware
//
#include <hip/hip_runtime.h>
#include <hip/hip_fp16.h>
#include <math.h>

typedef __attribute__((ext_vector_type(16))) _Float16 v16h;
typedef __attribute__((ext_vector_type(8)))  _Float16 v8h;
typedef __attribute__((ext_vector_type(8)))  float    v8f;

#define NN 8192
#define DD 1024
#define LSTR 40   // LDS row stride in halves (padded; keeps 16B alignment: 40%8==0)

// ---------------------------------------------------------------- fp32 -> f16
__global__ void cvt_kernel(const float* __restrict__ a, const float* __restrict__ b,
                           _Float16* __restrict__ ah, _Float16* __restrict__ bh, int n) {
  int i = blockIdx.x * blockDim.x + threadIdx.x;
  if (i < n) { ah[i] = (_Float16)a[i]; bh[i] = (_Float16)b[i]; }
}

// A fragment (16x32 f16): lane L holds row L%16; halves 0..7 = K[koff..koff+8),
// halves 8..15 = K[koff+16..koff+24), koff = (L/16)*8  (ISA 7.12.2 layout)
__device__ inline v16h load_fragA(const _Float16* s, int rb, int lane) {
  const _Float16* p = s + (rb + (lane & 15)) * LSTR + ((lane >> 4) * 8);
  v8h lo = *(const v8h*)p;
  v8h hi = *(const v8h*)(p + 16);
  v16h f;
#pragma unroll
  for (int x = 0; x < 8; ++x) { f[x] = lo[x]; f[x + 8] = hi[x]; }
  return f;
}

// B fragment (32x16 f16, B = T rows as columns): lane L holds column L%16;
// halves 0..7 = K[koff..koff+8), halves 8..15 = K[koff+16..koff+24), koff = (L/16)*8
__device__ inline v16h load_fragB(const _Float16* s, int rb, int lane) {
  const _Float16* p = s + (rb + (lane & 15)) * LSTR + ((lane >> 4) * 8);
  v8h lo = *(const v8h*)p;
  v8h hi = *(const v8h*)(p + 16);
  v16h f;
#pragma unroll
  for (int x = 0; x < 8; ++x) { f[x] = lo[x]; f[x + 8] = hi[x]; }
  return f;
}

// ------------------------------------------- C = I*T^T, P0 = f16(exp(C - 1))
__global__ void __launch_bounds__(256)
gemm_exp_kernel(const _Float16* __restrict__ A, const _Float16* __restrict__ B,
                _Float16* __restrict__ P) {
  __shared__ alignas(16) _Float16 lA[128 * LSTR];
  __shared__ alignas(16) _Float16 lB[128 * LSTR];
  const int t = threadIdx.x;
  const int lane = t & 31, wid = t >> 5;
  const int wm = (wid & 3) * 32;     // 4 waves along M (32 rows each)
  const int wn = (wid >> 2) * 64;    // 2 waves along N (64 cols each)
  const int bm = blockIdx.x * 128, bn = blockIdx.y * 128;

  v8f acc[2][4];
#pragma unroll
  for (int mi = 0; mi < 2; ++mi)
#pragma unroll
    for (int ni = 0; ni < 4; ++ni)
#pragma unroll
      for (int v = 0; v < 8; ++v) acc[mi][ni][v] = 0.0f;

  const int lr = t & 127;            // tile row this thread fills
  const int hb = (t >> 7) * 16;      // half-base within 32-wide K slab
  const _Float16* gA = A + (size_t)(bm + lr) * DD + hb;
  const _Float16* gB = B + (size_t)(bn + lr) * DD + hb;
  _Float16* sA = lA + lr * LSTR + hb;
  _Float16* sB = lB + lr * LSTR + hb;

  for (int k0 = 0; k0 < DD; k0 += 32) {
    v8h a0 = *(const v8h*)(gA + k0);
    v8h a1 = *(const v8h*)(gA + k0 + 8);
    v8h b0 = *(const v8h*)(gB + k0);
    v8h b1 = *(const v8h*)(gB + k0 + 8);
    if (k0 + 32 < DD) {              // next-slab prefetch -> global_prefetch_b8
      __builtin_prefetch(gA + k0 + 32, 0, 1);
      __builtin_prefetch(gB + k0 + 32, 0, 1);
    }
    __syncthreads();
    *(v8h*)sA = a0; *(v8h*)(sA + 8) = a1;
    *(v8h*)sB = b0; *(v8h*)(sB + 8) = b1;
    __syncthreads();

    v16h af[2], bf[4];
#pragma unroll
    for (int mi = 0; mi < 2; ++mi) af[mi] = load_fragA(lA, wm + mi * 16, lane);
#pragma unroll
    for (int ni = 0; ni < 4; ++ni) bf[ni] = load_fragB(lB, wn + ni * 16, lane);
#pragma unroll
    for (int mi = 0; mi < 2; ++mi)
#pragma unroll
      for (int ni = 0; ni < 4; ++ni)
        acc[mi][ni] = __builtin_amdgcn_wmma_f32_16x16x32_f16(
            false, af[mi], false, bf[ni], (short)0, acc[mi][ni], false, false);
  }

  // C/D layout: lane L, vgpr v -> M = v + 8*(L/16), N = L%16
  const int rowadd = 8 * (lane >> 4);
  const int col = lane & 15;
#pragma unroll
  for (int mi = 0; mi < 2; ++mi)
#pragma unroll
    for (int ni = 0; ni < 4; ++ni)
#pragma unroll
      for (int v = 0; v < 8; ++v) {
        int gm = bm + wm + mi * 16 + v + rowadd;
        int gn = bn + wn + ni * 16 + col;
        P[(size_t)gm * NN + gn] = (_Float16)__expf(acc[mi][ni][v] - 1.0f);
      }
}

// ----------------------------------------------------------- vector helpers
__global__ void init_vecs_kernel(float* r1, float* c1, float* r2, float* c2,
                                 float* v1, float* v2, float* u1, float* u2,
                                 float* m1, float* m2, float* s1, float* s2) {
  int i = blockIdx.x * 256 + threadIdx.x;
  if (i < NN) {
    r1[i] = c1[i] = r2[i] = c2[i] = 1.0f;
    v1[i] = v2[i] = u1[i] = u2[i] = 0.0f;
    m1[i] = m2[i] = 0.0f;
    s1[i] = s2[i] = 0.0f;
  }
}

__global__ void upd_r_kernel(float* r1, const float* v1, float* r2, const float* v2,
                             float* u1, float* u2) {
  int i = blockIdx.x * 256 + threadIdx.x;
  if (i < NN) { r1[i] = 1.0f / v1[i]; r2[i] = 1.0f / v2[i]; u1[i] = 0.0f; u2[i] = 0.0f; }
}

__global__ void upd_c_kernel(float* c1, const float* u1, float* c2, const float* u2,
                             float* v1, float* v2) {
  int i = blockIdx.x * 256 + threadIdx.x;
  if (i < NN) {
    const float bd = 0.1f * NN, bu = 0.9f * NN;
    float c = c1[i], u = u1[i];
    c *= fmaxf(bd / (u * c), 1.0f);
    c *= fminf(bu / (u * c), 1.0f);
    c1[i] = c;
    c = c2[i]; u = u2[i];
    c *= fmaxf(bd / (u * c), 1.0f);
    c *= fminf(bu / (u * c), 1.0f);
    c2[i] = c;
    v1[i] = 0.0f; v2[i] = 0.0f;
  }
}

// ---------- one read of P0 produces BOTH row-weighted and col-weighted sums
// rowOut[i] += sum_j P0[i][j]*colW[j];  colOut[j] += sum_i P0[i][j]*rowW[i]
__global__ void __launch_bounds__(256)
pass_rc_kernel(const _Float16* __restrict__ P, float* rowOut,
               const float* __restrict__ colW, float* colOut,
               const float* __restrict__ rowW) {
  const int t = threadIdx.x;
  const int j0 = blockIdx.x * 2048 + t * 8;
  const int i0 = blockIdx.y * 128;
  float cw[8], ca[8];
#pragma unroll
  for (int k = 0; k < 8; ++k) { cw[k] = colW[j0 + k]; ca[k] = 0.0f; }
  for (int r = 0; r < 128; ++r) {
    const int i = i0 + r;
    const float rw = rowW[i];
    v8h d = *(const v8h*)(P + (size_t)i * NN + j0);
    float rp = 0.0f;
#pragma unroll
    for (int k = 0; k < 8; ++k) {
      float p = (float)d[k];
      rp = fmaf(p, cw[k], rp);
      ca[k] = fmaf(p, rw, ca[k]);
    }
#pragma unroll
    for (int off = 16; off; off >>= 1) rp += __shfl_down(rp, off, 32);
    if ((t & 31) == 0) atomicAdd(&rowOut[i], rp);
  }
#pragma unroll
  for (int k = 0; k < 8; ++k) atomicAdd(&colOut[j0 + k], ca[k]);
}

// ---------------- max pass for stable logsumexp (values > 0 -> uint ordering)
__global__ void __launch_bounds__(256)
pass_max_kernel(const _Float16* __restrict__ P, float* rowM,
                const float* __restrict__ colW, float* colM,
                const float* __restrict__ rowW) {
  const int t = threadIdx.x;
  const int j0 = blockIdx.x * 2048 + t * 8;
  const int i0 = blockIdx.y * 128;
  float cw[8], cm[8];
#pragma unroll
  for (int k = 0; k < 8; ++k) { cw[k] = colW[j0 + k]; cm[k] = 0.0f; }
  for (int r = 0; r < 128; ++r) {
    const int i = i0 + r;
    const float rw = rowW[i];
    v8h d = *(const v8h*)(P + (size_t)i * NN + j0);
    float rm = 0.0f;
#pragma unroll
    for (int k = 0; k < 8; ++k) {
      float p = (float)d[k];
      rm = fmaxf(rm, p * cw[k]);
      cm[k] = fmaxf(cm[k], p * rw);
    }
#pragma unroll
    for (int off = 16; off; off >>= 1) rm = fmaxf(rm, __shfl_down(rm, off, 32));
    if ((t & 31) == 0) atomicMax((unsigned int*)&rowM[i], __float_as_uint(rm));
  }
#pragma unroll
  for (int k = 0; k < 8; ++k)
    atomicMax((unsigned int*)&colM[j0 + k], __float_as_uint(cm[k]));
}

// ---- sumexp pass: SE1[i] += exp(r1_i*c1_j*P - r1_i*M1_i), SE2[j] likewise
__global__ void __launch_bounds__(256)
pass_sumexp_kernel(const _Float16* __restrict__ P,
                   float* SE1, const float* __restrict__ c1,
                   const float* __restrict__ r1, const float* __restrict__ M1,
                   float* SE2, const float* __restrict__ c2,
                   const float* __restrict__ r2, const float* __restrict__ M2) {
  const int t = threadIdx.x;
  const int j0 = blockIdx.x * 2048 + t * 8;
  const int i0 = blockIdx.y * 128;
  float c1r[8], r2r[8], m2r[8], ca[8];
#pragma unroll
  for (int k = 0; k < 8; ++k) {
    c1r[k] = c1[j0 + k];
    r2r[k] = r2[j0 + k];
    m2r[k] = r2r[k] * M2[j0 + k];
    ca[k] = 0.0f;
  }
  for (int r = 0; r < 128; ++r) {
    const int i = i0 + r;
    const float r1i = r1[i], c2i = c2[i];
    const float m1 = r1i * M1[i];
    v8h d = *(const v8h*)(P + (size_t)i * NN + j0);
    float rp = 0.0f;
#pragma unroll
    for (int k = 0; k < 8; ++k) {
      float p = (float)d[k];
      rp += __expf(fmaf(r1i * c1r[k], p, -m1));
      ca[k] += __expf(fmaf(r2r[k] * c2i, p, -m2r[k]));
    }
#pragma unroll
    for (int off = 16; off; off >>= 1) rp += __shfl_down(rp, off, 32);
    if ((t & 31) == 0) atomicAdd(&SE1[i], rp);
  }
#pragma unroll
  for (int k = 0; k < 8; ++k) atomicAdd(&SE2[j0 + k], ca[k]);
}

// ------------------------- deterministic final reduction (single workgroup)
__global__ void __launch_bounds__(1024)
final_loss_kernel(const _Float16* __restrict__ P, const int* __restrict__ labels,
                  const float* r1, const float* c1, const float* m1, const float* s1,
                  const float* r2, const float* c2, const float* m2, const float* s2,
                  float* out) {
  __shared__ float red[1024];
  const int t = threadIdx.x;
  float acc = 0.0f;
  for (int b = 0; b < 8; ++b) {
    int i = t + b * 1024;
    int li = labels[i];
    float lse1 = r1[i] * m1[i] + __logf(s1[i]);
    float p1 = r1[i] * c1[li] * (float)P[(size_t)i * NN + li];
    float lse2 = r2[i] * m2[i] + __logf(s2[i]);
    float p2 = r2[i] * c2[li] * (float)P[(size_t)li * NN + i];
    acc += (lse1 - p1) + (lse2 - p2);
  }
  red[t] = acc;
  __syncthreads();
  for (int s = 512; s; s >>= 1) {
    if (t < s) red[t] += red[t + s];
    __syncthreads();
  }
  if (t == 0) out[0] = red[0] / (2.0f * NN);
}

extern "C" void kernel_launch(void* const* d_in, const int* in_sizes, int n_in,
                              void* d_out, int out_size, void* d_ws, size_t ws_size,
                              hipStream_t stream) {
  const float* imgF = (const float*)d_in[0];
  const float* txtF = (const float*)d_in[1];
  const int* labels = (const int*)d_in[3];  // d_in[2] = logit_scale (unused by ref math)
  float* out = (float*)d_out;

  char* ws = (char*)d_ws;
  _Float16* P0 = (_Float16*)ws;                               // 128 MB
  _Float16* Ih = (_Float16*)(ws + (size_t)NN * NN * 2);       // 16 MB
  _Float16* Th = Ih + (size_t)NN * DD;                        // 16 MB
  float* fv = (float*)(Th + (size_t)NN * DD);
  float *r1 = fv,            *c1 = fv + NN,     *r2 = fv + 2 * NN, *c2 = fv + 3 * NN;
  float *v1 = fv + 4 * NN,   *v2 = fv + 5 * NN, *u1 = fv + 6 * NN, *u2 = fv + 7 * NN;
  float *m1 = fv + 8 * NN,   *m2 = fv + 9 * NN, *s1 = fv + 10 * NN, *s2 = fv + 11 * NN;

  cvt_kernel<<<(NN * DD + 255) / 256, 256, 0, stream>>>(imgF, txtF, Ih, Th, NN * DD);
  gemm_exp_kernel<<<dim3(64, 64), 256, 0, stream>>>(Ih, Th, P0);
  init_vecs_kernel<<<NN / 256, 256, 0, stream>>>(r1, c1, r2, c2, v1, v2, u1, u2,
                                                 m1, m2, s1, s2);
  for (int it = 0; it < 5; ++it) {
    // v1 = P0*c1 (loss1 rowsum core), v2 = P0^T*c2 (loss2 rowsum core)
    pass_rc_kernel<<<dim3(4, 64), 256, 0, stream>>>(P0, v1, c1, v2, c2);
    upd_r_kernel<<<NN / 256, 256, 0, stream>>>(r1, v1, r2, v2, u1, u2);
    // u2 = P0*r2, u1 = P0^T*r1 (colsum cores; reused for both max/min steps)
    pass_rc_kernel<<<dim3(4, 64), 256, 0, stream>>>(P0, u2, r2, u1, r1);
    upd_c_kernel<<<NN / 256, 256, 0, stream>>>(c1, u1, c2, u2, v1, v2);
  }
  pass_max_kernel<<<dim3(4, 64), 256, 0, stream>>>(P0, m1, c1, m2, c2);
  pass_sumexp_kernel<<<dim3(4, 64), 256, 0, stream>>>(P0, s1, c1, r1, m1,
                                                      s2, c2, r2, m2);
  final_loss_kernel<<<1, 1024, 0, stream>>>(P0, labels, r1, c1, m1, s1,
                                            r2, c2, m2, s2, out);
}
